// JiT_59665685676695
// MI455X (gfx1250) — 16-bit and 8-bit matrix operand order repaired and checked by reading, compiled, not run on hardware
//
#include <hip/hip_runtime.h>
#include <math.h>

typedef __attribute__((ext_vector_type(16))) _Float16 v16h;
typedef __attribute__((ext_vector_type(8)))  float    v8f;
typedef __attribute__((ext_vector_type(8)))  int      v8i;

#define Bx     32
#define IMGSZ  256
#define PATCH  16
#define CIN    3
#define HID    768
#define DEPTHN 4
#define HEADS  12
#define HDIM   64
#define NTOK   256
#define TOK    (Bx * NTOK)
#define FFNH   2048
#define TFREQ  256

__device__ __forceinline__ int iabs_(int v) { return v < 0 ? -v : v; }
__device__ __forceinline__ int imax_(int a, int b) { return a > b ? a : b; }
__device__ __forceinline__ float silu_(float v) { return v / (1.f + __expf(-v)); }

__device__ __forceinline__ unsigned lds_off_(const void* p) {
  return (unsigned)(size_t)(__attribute__((address_space(3))) const char*)p;
}
__device__ __forceinline__ void async_g2l_b128(unsigned lds_byte_off, const void* gaddr) {
  asm volatile("global_load_async_to_lds_b128 %0, %1, off"
               :: "v"(lds_byte_off), "v"(gaddr) : "memory");
}
__device__ __forceinline__ void wait_async0_() {
  asm volatile("s_wait_asynccnt 0x0" ::: "memory");
}

#define GBM 64
#define GBN 32
#define GBK 32
#define GLDA 36

__global__ __launch_bounds__(256)
void gemm_f16(const float* __restrict__ Ab, const float* __restrict__ Wb,
              const float* __restrict__ bias, float* __restrict__ Cb,
              int M, int Nn, int K,
              long sA, long sW, long sC, float alpha, int act)
{
  const float* A = Ab + (long)blockIdx.z * sA;
  const float* W = Wb + (long)blockIdx.z * sW;
  float*       C = Cb + (long)blockIdx.z * sC;

  __shared__ _Float16 As[GBM][GLDA];
  __shared__ _Float16 Bs[GBN][GLDA];

  const int tid  = threadIdx.x;
  const int lane = tid & 31;
  const int wave = tid >> 5;
  const int wm   = wave & 3;
  const int wn   = wave >> 2;
  const int m0   = blockIdx.y * GBM;
  const int n0   = blockIdx.x * GBN;
  const int half = lane >> 4;
  const int lr   = lane & 15;

  const int sr = tid >> 3;
  const int sc = tid & 7;

  v8f acc = {};

  for (int k0 = 0; k0 < K; k0 += GBK) {
#pragma unroll
    for (int it = 0; it < 2; ++it) {
      int r = sr + it * 32;
      int gm = m0 + r; gm = gm < M ? gm : M - 1;
      float4 v4 = *(const float4*)(A + (long)gm * K + k0 + sc * 4);
      union { _Float16 h[4]; unsigned long long u; } pk;
      pk.h[0] = (_Float16)v4.x; pk.h[1] = (_Float16)v4.y;
      pk.h[2] = (_Float16)v4.z; pk.h[3] = (_Float16)v4.w;
      *(unsigned long long*)&As[r][sc * 4] = pk.u;
    }
    {
      int gn = n0 + sr; gn = gn < Nn ? gn : Nn - 1;
      float4 v4 = *(const float4*)(W + (long)gn * K + k0 + sc * 4);
      union { _Float16 h[4]; unsigned long long u; } pk;
      pk.h[0] = (_Float16)v4.x; pk.h[1] = (_Float16)v4.y;
      pk.h[2] = (_Float16)v4.z; pk.h[3] = (_Float16)v4.w;
      *(unsigned long long*)&Bs[sr][sc * 4] = pk.u;
    }
    __syncthreads();

    if (k0 + GBK < K) {
      __builtin_prefetch(A + (long)(m0 + wave * 8) * K + (k0 + GBK), 0, 1);
    }

    union { v16h v; unsigned u[8]; } fa, fb;
    {
      const unsigned* pr = (const unsigned*)&As[wm * 16 + lr][0];
      const int kb = half * 8;
#pragma unroll
      for (int i = 0; i < 4; ++i) {
        fa.u[i]     = pr[(kb >> 1) + i];
        fa.u[i + 4] = pr[((kb + 16) >> 1) + i];
      }
      const unsigned* pc = (const unsigned*)&Bs[wn * 16 + lr][0];
#pragma unroll
      for (int i = 0; i < 4; ++i) { fb.u[i] = pc[half * 4 + i]; fb.u[i + 4] = pc[8 + half * 4 + i]; }
    }
    acc = __builtin_amdgcn_wmma_f32_16x16x32_f16(false, fa.v, false, fb.v,
                                                 (short)0, acc, false, false);
    __syncthreads();
  }

  const int col = n0 + wn * 16 + lr;
  const float bv = (bias != nullptr && col < Nn) ? bias[col] : 0.f;
#pragma unroll
  for (int r = 0; r < 8; ++r) {
    int row = m0 + wm * 16 + (half ? r + 8 : r);
    if (row < M && col < Nn) {
      float v = acc[r] * alpha + bv;
      if (act == 1) v = silu_(v);
      C[(long)row * Nn + col] = v;
    }
  }
}

#define QBM 64
#define QBN 32
#define QBK 64
#define QLDA 80

__device__ __forceinline__ v8i fragA_i8(const signed char* rowp, int off) {
  const int kb[8] = {0, 4, 16, 20, 32, 36, 48, 52};
  const int* p = (const int*)rowp;
  union { v8i v; int d[8]; } r;
#pragma unroll
  for (int i = 0; i < 8; ++i) r.d[i] = p[(kb[i] + off) >> 2];
  return r.v;
}
__device__ __forceinline__ v8i fragB_i8(const signed char* rowp, int half) {
  const int* p = (const int*)rowp;
  union { v8i v; int d[8]; } r;
#pragma unroll
  for (int i = 0; i < 8; ++i) r.d[i] = p[half * 2 + (i >> 1) * 4 + (i & 1)];
  return r.v;
}

__global__ __launch_bounds__(256)
void gemm_iu8_dual(const signed char* __restrict__ Amsb,
                   const signed char* __restrict__ Alsb,
                   const signed char* __restrict__ Wq,
                   int* __restrict__ Cmsb, int* __restrict__ Clsb,
                   int M, int Nn, int K, int* maxM, int* maxL)
{
  __shared__ signed char AsM[QBM][QLDA];
  __shared__ signed char AsL[QBM][QLDA];
  __shared__ signed char Ws [QBN][QLDA];

  const int tid  = threadIdx.x;
  const int lane = tid & 31;
  const int wave = tid >> 5;
  const int wm   = wave & 3;
  const int wn   = wave >> 2;
  const int m0   = blockIdx.y * QBM;
  const int n0   = blockIdx.x * QBN;
  const int half = lane >> 4;
  const int lr   = lane & 15;

  const int ar = tid >> 2;
  const int ac = (tid & 3) * 16;

  v8i accM = {}, accL = {};

  for (int k0 = 0; k0 < K; k0 += QBK) {
    async_g2l_b128(lds_off_(&AsM[ar][ac]), Amsb + (long)(m0 + ar) * K + k0 + ac);
    async_g2l_b128(lds_off_(&AsL[ar][ac]), Alsb + (long)(m0 + ar) * K + k0 + ac);
    if (tid < 128) {
      int wr = tid >> 2;
      async_g2l_b128(lds_off_(&Ws[wr][ac]), Wq + (long)(n0 + wr) * K + k0 + ac);
    }
    wait_async0_();
    __syncthreads();

    if (k0 + QBK < K) {
      __builtin_prefetch(Amsb + (long)(m0 + wave * 8) * K + (k0 + QBK), 0, 1);
    }

    v8i fm = fragA_i8(&AsM[wm * 16 + lr][0], half * 8);
    v8i fl = fragA_i8(&AsL[wm * 16 + lr][0], half * 8);
    v8i fb = fragB_i8(&Ws [wn * 16 + lr][0], half);
    accM = __builtin_amdgcn_wmma_i32_16x16x64_iu8(true,  fm, true, fb, accM, false, false);
    accL = __builtin_amdgcn_wmma_i32_16x16x64_iu8(false, fl, true, fb, accL, false, false);
    __syncthreads();
  }

  const int col = n0 + wn * 16 + lr;
  int mm = 0, ml = 0;
#pragma unroll
  for (int r = 0; r < 8; ++r) {
    int row = m0 + wm * 16 + (half ? r + 8 : r);
    int vM = accM[r], vL = accL[r];
    Cmsb[(long)row * Nn + col] = vM;
    Clsb[(long)row * Nn + col] = vL;
    mm = imax_(mm, iabs_(vM));
    ml = imax_(ml, iabs_(vL));
  }
  __shared__ int rb[256];
  rb[tid] = mm; __syncthreads();
  for (int s = 128; s > 0; s >>= 1) { if (tid < s) rb[tid] = imax_(rb[tid], rb[tid + s]); __syncthreads(); }
  if (tid == 0) atomicMax(maxM, rb[0]);
  __syncthreads();
  rb[tid] = ml; __syncthreads();
  for (int s = 128; s > 0; s >>= 1) { if (tid < s) rb[tid] = imax_(rb[tid], rb[tid + s]); __syncthreads(); }
  if (tid == 0) atomicMax(maxL, rb[0]);
}

__global__ void zero_slots(unsigned* p, int n) { if ((int)threadIdx.x < n) p[threadIdx.x] = 0u; }

__global__ __launch_bounds__(256)
void reduce_absmax_f32(const float* __restrict__ x, long n, unsigned* slot) {
  __shared__ float sm[256];
  float m = 0.f;
  for (long i = (long)blockIdx.x * 256 + threadIdx.x; i < n; i += (long)gridDim.x * 256)
    m = fmaxf(m, fabsf(x[i]));
  sm[threadIdx.x] = m; __syncthreads();
  for (int s = 128; s > 0; s >>= 1) { if ((int)threadIdx.x < s) sm[threadIdx.x] = fmaxf(sm[threadIdx.x], sm[threadIdx.x + s]); __syncthreads(); }
  if (threadIdx.x == 0) atomicMax(slot, __float_as_uint(sm[0]));
}

__global__ __launch_bounds__(256)
void quantize_w(const float* __restrict__ W, signed char* __restrict__ Wq, long n,
                const unsigned* wmax_slot) {
  float ws = fmaxf(__uint_as_float(*wmax_slot) / 127.f, 1e-8f);
  for (long i = (long)blockIdx.x * 256 + threadIdx.x; i < n; i += (long)gridDim.x * 256) {
    float q = fminf(fmaxf(rintf(W[i] / ws), -128.f), 127.f);
    Wq[i] = (signed char)q;
  }
}

__global__ __launch_bounds__(256)
void quantize_act(const float* __restrict__ x, signed char* __restrict__ msb,
                  signed char* __restrict__ lsb, long n, const unsigned* amax_slot) {
  float as = fmaxf(__uint_as_float(*amax_slot) / 2047.f, 1e-8f);
  for (long i = (long)blockIdx.x * 256 + threadIdx.x; i < n; i += (long)gridDim.x * 256) {
    float xi = fminf(fmaxf(rintf(x[i] / as), -2048.f), 2047.f);
    float xm = fminf(fmaxf(floorf(xi / 16.f), -128.f), 127.f);
    float xl = fminf(fmaxf(xi - xm * 16.f, 0.f), 15.f);
    msb[i] = (signed char)xm;
    lsb[i] = (signed char)xl;
  }
}

__global__ __launch_bounds__(256)
void bitserial_finalize(const int* Cm, const int* Cl, const float* __restrict__ bias,
                        float* out, float* resid, const float* __restrict__ mbuf, int goff,
                        long M, int Nn,
                        const unsigned* xs_slot, const unsigned* ws_slot,
                        const int* ym_slot, const int* yl_slot)
{
  float act_s = fmaxf(__uint_as_float(*xs_slot) / 2047.f, 1e-8f);
  float w_s   = fmaxf(__uint_as_float(*ws_slot) / 127.f, 1e-8f);
  float g  = act_s * w_s * 16.f;
  float sm = (float)(*ym_slot) / 511.f + 1e-8f;
  float sl = (float)(*yl_slot) / 511.f + 1e-8f;
  long total = M * Nn;
  for (long i = (long)blockIdx.x * 256 + threadIdx.x; i < total; i += (long)gridDim.x * 256) {
    long row = i / Nn; int col = (int)(i - row * Nn);
    float am = fminf(fmaxf(rintf((float)Cm[i] / sm), -512.f), 511.f) * sm;
    float al = fminf(fmaxf(rintf((float)Cl[i] / sl), -512.f), 511.f) * sl;
    float y = am * g + al * (g / 16.f) + bias[col];
    if (resid) {
      int b = (int)(row / NTOK);
      resid[i] += mbuf[(long)b * (6 * HID) + goff + col] * y;
    } else {
      out[i] = y;
    }
  }
}

__global__ __launch_bounds__(256)
void patchify(const float* __restrict__ x, float* __restrict__ out) {
  long i = (long)blockIdx.x * 256 + threadIdx.x;
  if (i >= (long)TOK * 768) return;
  int k = (int)(i % 768); long tok = i / 768;
  int n = (int)(tok % NTOK); int b = (int)(tok / NTOK);
  int c = k / 256, rem = k % 256, pi = rem / 16, pj = rem % 16;
  int ti = n / 16, tj = n % 16;
  out[i] = x[(((long)b * CIN + c) * IMGSZ + (ti * 16 + pi)) * IMGSZ + (tj * 16 + pj)];
}

__global__ void tstep_embed(const float* __restrict__ t, float* __restrict__ emb) {
  int b = blockIdx.x, j = threadIdx.x;
  float fr = __expf(-9.210340371976184f * (float)j / 128.f);
  float a = t[b] * fr;
  emb[b * TFREQ + j]       = __cosf(a);
  emb[b * TFREQ + 128 + j] = __sinf(a);
}

__global__ __launch_bounds__(256)
void add_label_silu(const float* __restrict__ t2, const float* __restrict__ lab,
                    const int* __restrict__ y, float* __restrict__ c,
                    float* __restrict__ csilu) {
  int b = blockIdx.x;
  for (int k = threadIdx.x; k < HID; k += 256) {
    float v = t2[(long)b * HID + k] + lab[(long)y[b] * HID + k];
    c[(long)b * HID + k] = v;
    csilu[(long)b * HID + k] = silu_(v);
  }
}

__global__ __launch_bounds__(256)
void rms_modulate(const float* __restrict__ x, const float* __restrict__ w,
                  const float* __restrict__ mbuf, int mstride, int sh_off, int sc_off,
                  float* __restrict__ out) {
  long tok = blockIdx.x;
  int b = (int)(tok / NTOK);
  const float* xr = x + tok * HID;
  __shared__ float red[256];
  float ss = 0.f;
  for (int k = threadIdx.x; k < HID; k += 256) { float v = xr[k]; ss += v * v; }
  red[threadIdx.x] = ss; __syncthreads();
  for (int s = 128; s > 0; s >>= 1) { if ((int)threadIdx.x < s) red[threadIdx.x] += red[threadIdx.x + s]; __syncthreads(); }
  float inv = rsqrtf(red[0] / (float)HID + 1e-6f);
  const float* mb = mbuf ? (mbuf + (long)b * mstride) : nullptr;
  for (int k = threadIdx.x; k < HID; k += 256) {
    float v = xr[k] * inv * w[k];
    if (mb) v = v * (1.f + mb[sc_off + k]) + mb[sh_off + k];
    out[tok * HID + k] = v;
  }
}

__global__ __launch_bounds__(64)
void qk_prep(const float* __restrict__ qkv, const float* __restrict__ qn_w,
             const float* __restrict__ kn_w, float* __restrict__ qb,
             float* __restrict__ kb, float* __restrict__ vT) {
  int idx = blockIdx.x;
  int h = idx % HEADS, n = (idx / HEADS) % NTOK, b = idx / (HEADS * NTOK);
  int d = threadIdx.x;
  const float* base = qkv + (long)(b * NTOK + n) * (3 * HID);
  float q = base[h * HDIM + d];
  float k = base[HID + h * HDIM + d];
  float v = base[2 * HID + h * HDIM + d];
  __shared__ float red[64];
  red[d] = q * q; __syncthreads();
  for (int s = 32; s > 0; s >>= 1) { if (d < s) red[d] += red[d + s]; __syncthreads(); }
  float qn = q * rsqrtf(red[0] / (float)HDIM + 1e-6f) * qn_w[d];
  __syncthreads();
  red[d] = k * k; __syncthreads();
  for (int s = 32; s > 0; s >>= 1) { if (d < s) red[d] += red[d + s]; __syncthreads(); }
  float kn = k * rsqrtf(red[0] / (float)HDIM + 1e-6f) * kn_w[d];
  __shared__ float qs[64], ks[64];
  __syncthreads();
  qs[d] = qn; ks[d] = kn; __syncthreads();
  int p2 = d >> 1;
  float fr = __expf(-9.210340371976184f * (float)p2 / 32.f);
  float ang = (float)n * fr, co = __cosf(ang), si = __sinf(ang);
  float x1 = qs[2 * p2], x2 = qs[2 * p2 + 1];
  float qo = (d & 1) ? (x1 * si + x2 * co) : (x1 * co - x2 * si);
  float y1 = ks[2 * p2], y2 = ks[2 * p2 + 1];
  float ko = (d & 1) ? (y1 * si + y2 * co) : (y1 * co - y2 * si);
  long hb = (long)(b * HEADS + h) * NTOK;
  qb[(hb + n) * HDIM + d] = qo;
  kb[(hb + n) * HDIM + d] = ko;
  vT[((long)(b * HEADS + h) * HDIM + d) * NTOK + n] = v;
}

__global__ __launch_bounds__(256)
void softmax256(float* __restrict__ S) {
  long row = blockIdx.x;
  float* r = S + row * 256;
  __shared__ float red[256];
  float v = r[threadIdx.x];
  red[threadIdx.x] = v; __syncthreads();
  for (int s = 128; s > 0; s >>= 1) { if ((int)threadIdx.x < s) red[threadIdx.x] = fmaxf(red[threadIdx.x], red[threadIdx.x + s]); __syncthreads(); }
  float mx = red[0]; __syncthreads();
  float e = __expf(v - mx);
  red[threadIdx.x] = e; __syncthreads();
  for (int s = 128; s > 0; s >>= 1) { if ((int)threadIdx.x < s) red[threadIdx.x] += red[threadIdx.x + s]; __syncthreads(); }
  r[threadIdx.x] = e / red[0];
}

__global__ __launch_bounds__(256)
void o_merge(const float* __restrict__ ob, float* __restrict__ oc) {
  long i = (long)blockIdx.x * 256 + threadIdx.x;
  if (i >= (long)TOK * HID) return;
  int k = (int)(i % HID); long tok = i / HID;
  int n = (int)(tok % NTOK), b = (int)(tok / NTOK);
  int h = k / HDIM, d = k % HDIM;
  oc[i] = ob[((long)(b * HEADS + h) * NTOK + n) * HDIM + d];
}

__global__ __launch_bounds__(256)
void residual_gated(float* __restrict__ hx, const float* __restrict__ yv,
                    const float* __restrict__ mbuf, int goff) {
  long i = (long)blockIdx.x * 256 + threadIdx.x;
  if (i >= (long)TOK * HID) return;
  int col = (int)(i % HID); long tok = i / HID; int b = (int)(tok / NTOK);
  hx[i] += mbuf[(long)b * (6 * HID) + goff + col] * yv[i];
}

__global__ __launch_bounds__(256)
void swiglu(const float* __restrict__ h12, float* __restrict__ act) {
  long i = (long)blockIdx.x * 256 + threadIdx.x;
  if (i >= (long)TOK * FFNH) return;
  long tok = i / FFNH; int j = (int)(i % FFNH);
  float a = h12[tok * (2 * FFNH) + j];
  float b2 = h12[tok * (2 * FFNH) + FFNH + j];
  act[i] = silu_(a) * b2;
}

__global__ __launch_bounds__(256)
void unpatchify(const float* __restrict__ hf, float* __restrict__ img) {
  long i = (long)blockIdx.x * 256 + threadIdx.x;
  if (i >= (long)Bx * CIN * IMGSZ * IMGSZ) return;
  int iw = (int)(i % IMGSZ); long r = i / IMGSZ;
  int ih = (int)(r % IMGSZ); long r2 = r / IMGSZ;
  int c = (int)(r2 % CIN); int b = (int)(r2 / CIN);
  int ti = ih / 16, pi = ih % 16, tj = iw / 16, pj = iw % 16;
  int n = ti * 16 + tj;
  int k = (pi * 16 + pj) * 3 + c;
  img[i] = hf[(long)(b * NTOK + n) * 768 + k];
}

extern "C" void kernel_launch(void* const* d_in, const int* in_sizes, int n_in,
                              void* d_out, int out_size, void* d_ws, size_t ws_size,
                              hipStream_t stream) {
  (void)in_sizes; (void)n_in; (void)out_size; (void)ws_size;
  auto F = [&](int i) { return (const float*)d_in[i]; };
  const float* xin = F(0);
  const float* tin = F(1);
  const int*   yin = (const int*)d_in[2];
  const float *ada_b[DEPTHN], *ada_w[DEPTHN], *kn_w[DEPTHN], *n1_w[DEPTHN], *n2_w[DEPTHN],
              *proj_b[DEPTHN], *proj_w[DEPTHN], *qkv_b[DEPTHN], *qkv_w[DEPTHN],
              *qn_w[DEPTHN], *w12_b[DEPTHN], *w12_w[DEPTHN], *w3_b[DEPTHN], *w3_w[DEPTHN];
  for (int i = 0; i < DEPTHN; ++i) {
    ada_b[i] = F(3 + i);  ada_w[i] = F(7 + i);  kn_w[i] = F(16 + i);
    n1_w[i] = F(21 + i);  n2_w[i] = F(25 + i);
    proj_b[i] = F(32 + i); proj_w[i] = F(36 + i);
    qkv_b[i] = F(40 + i);  qkv_w[i] = F(44 + i); qn_w[i] = F(48 + i);
    w12_b[i] = F(56 + i);  w12_w[i] = F(60 + i);
    w3_b[i] = F(64 + i);   w3_w[i] = F(68 + i);
  }
  const float* fada_b = F(11); const float* fada_w = F(12);
  const float* flin_b = F(13); const float* flin_w = F(14);
  const float* fn_w   = F(15); const float* label_tab = F(20);
  const float* pe1_w  = F(29); const float* pe2_b = F(30); const float* pe2_w = F(31);
  const float* t_b1   = F(52); const float* t_b2  = F(53);
  const float* t_w1   = F(54); const float* t_w2  = F(55);

  char* ws = (char*)d_ws;
  size_t off = 0;
  auto alloc = [&](size_t bytes) { size_t r = off; off += (bytes + 255) & ~(size_t)255; return r; };
  float* hbuf  = (float*)(ws + alloc((size_t)TOK * HID * 4));
  float* cbuf  = (float*)(ws + alloc((size_t)Bx * HID * 4));
  float* csilu = (float*)(ws + alloc((size_t)Bx * HID * 4));
  float* emb   = (float*)(ws + alloc((size_t)Bx * TFREQ * 4));
  float* t1    = (float*)(ws + alloc((size_t)Bx * HID * 4));
  float* mb    = (float*)(ws + alloc((size_t)Bx * 6 * HID * 4));
  float* xm    = (float*)(ws + alloc((size_t)TOK * HID * 4));
  float* tmp   = (float*)(ws + alloc((size_t)TOK * HID * 4));
  signed char* wq = (signed char*)(ws + alloc((size_t)2 * FFNH * HID));
  unsigned* slots = (unsigned*)(ws + alloc(64 * 4));
  char* S = ws + alloc(0);
  float* qkv  = (float*)(S);
  float* qb   = (float*)(S + (size_t)TOK * 3 * HID * 4);
  float* kbf  = qb  + (size_t)Bx * HEADS * NTOK * HDIM;
  float* vT   = kbf + (size_t)Bx * HEADS * NTOK * HDIM;
  float* Sc   = vT  + (size_t)Bx * HEADS * NTOK * HDIM;
  float* ob   = Sc  + (size_t)Bx * HEADS * NTOK * NTOK;
  float* ocat = ob  + (size_t)Bx * HEADS * NTOK * HDIM;
  signed char* amsb = (signed char*)(S);
  signed char* alsb = amsb + (size_t)TOK * FFNH;
  int* cmsb = (int*)(alsb + (size_t)TOK * FFNH);
  int* clsb = cmsb + (size_t)TOK * 2 * FFNH;

  auto gemm = [&](const float* A, const float* W, const float* bias, float* C,
                  int M, int Nn, int K, int batch, long sA, long sW, long sC,
                  float alpha, int act) {
    dim3 g((Nn + GBN - 1) / GBN, (M + GBM - 1) / GBM, batch);
    gemm_f16<<<g, 256, 0, stream>>>(A, W, bias, C, M, Nn, K, sA, sW, sC, alpha, act);
  };
  auto gs = [&](long n) { long b = (n + 255) / 256; return (int)(b > 4096 ? 4096 : b); };

  float* patches = qkv;
  patchify<<<(int)(((long)TOK * 768 + 255) / 256), 256, 0, stream>>>(xin, patches);
  gemm(patches, pe1_w, nullptr, tmp, TOK, 768, 768, 1, 0, 0, 0, 1.f, 0);
  gemm(tmp, pe2_w, pe2_b, hbuf, TOK, HID, 768, 1, 0, 0, 0, 1.f, 0);

  tstep_embed<<<Bx, 128, 0, stream>>>(tin, emb);
  gemm(emb, t_w1, t_b1, t1, Bx, HID, TFREQ, 1, 0, 0, 0, 1.f, 1);
  gemm(t1, t_w2, t_b2, tmp, Bx, HID, HID, 1, 0, 0, 0, 1.f, 0);
  add_label_silu<<<Bx, 256, 0, stream>>>(tmp, label_tab, yin, cbuf, csilu);

  const int BH = Bx * HEADS;
  for (int i = 0; i < DEPTHN; ++i) {
    gemm(csilu, ada_w[i], ada_b[i], mb, Bx, 6 * HID, HID, 1, 0, 0, 0, 1.f, 0);

    rms_modulate<<<TOK, 256, 0, stream>>>(hbuf, n1_w[i], mb, 6 * HID, 0, HID, xm);
    gemm(xm, qkv_w[i], qkv_b[i], qkv, TOK, 3 * HID, HID, 1, 0, 0, 0, 1.f, 0);
    qk_prep<<<Bx * NTOK * HEADS, 64, 0, stream>>>(qkv, qn_w[i], kn_w[i], qb, kbf, vT);
    gemm(qb, kbf, nullptr, Sc, NTOK, NTOK, HDIM, BH,
         (long)NTOK * HDIM, (long)NTOK * HDIM, (long)NTOK * NTOK, 0.125f, 0);
    softmax256<<<BH * NTOK, 256, 0, stream>>>(Sc);
    gemm(Sc, vT, nullptr, ob, NTOK, HDIM, NTOK, BH,
         (long)NTOK * NTOK, (long)HDIM * NTOK, (long)NTOK * HDIM, 1.f, 0);
    o_merge<<<(int)(((long)TOK * HID + 255) / 256), 256, 0, stream>>>(ob, ocat);
    gemm(ocat, proj_w[i], proj_b[i], tmp, TOK, HID, HID, 1, 0, 0, 0, 1.f, 0);
    residual_gated<<<(int)(((long)TOK * HID + 255) / 256), 256, 0, stream>>>(hbuf, tmp, mb, 2 * HID);

    rms_modulate<<<TOK, 256, 0, stream>>>(hbuf, n2_w[i], mb, 6 * HID, 3 * HID, 4 * HID, xm);
    zero_slots<<<1, 64, 0, stream>>>(slots, 8);
    reduce_absmax_f32<<<gs((long)TOK * HID), 256, 0, stream>>>(xm, (long)TOK * HID, slots + 0);
    reduce_absmax_f32<<<gs((long)2 * FFNH * HID), 256, 0, stream>>>(w12_w[i], (long)2 * FFNH * HID, slots + 1);
    quantize_act<<<gs((long)TOK * HID), 256, 0, stream>>>(xm, amsb, alsb, (long)TOK * HID, slots + 0);
    quantize_w<<<gs((long)2 * FFNH * HID), 256, 0, stream>>>(w12_w[i], wq, (long)2 * FFNH * HID, slots + 1);
    {
      dim3 g((2 * FFNH) / QBN, TOK / QBM, 1);
      gemm_iu8_dual<<<g, 256, 0, stream>>>(amsb, alsb, wq, cmsb, clsb,
                                           TOK, 2 * FFNH, HID,
                                           (int*)(slots + 2), (int*)(slots + 3));
    }
    bitserial_finalize<<<gs((long)TOK * 2 * FFNH), 256, 0, stream>>>(
        cmsb, clsb, w12_b[i], (float*)cmsb, nullptr, nullptr, 0,
        TOK, 2 * FFNH, slots + 0, slots + 1, (const int*)(slots + 2), (const int*)(slots + 3));
    swiglu<<<(int)(((long)TOK * FFNH + 255) / 256), 256, 0, stream>>>((float*)cmsb, (float*)clsb);
    float* actb = (float*)clsb;
    int* cm2 = cmsb;
    int* cl2 = cmsb + (size_t)TOK * HID;
    reduce_absmax_f32<<<gs((long)TOK * FFNH), 256, 0, stream>>>(actb, (long)TOK * FFNH, slots + 4);
    reduce_absmax_f32<<<gs((long)HID * FFNH), 256, 0, stream>>>(w3_w[i], (long)HID * FFNH, slots + 5);
    quantize_act<<<gs((long)TOK * FFNH), 256, 0, stream>>>(actb, amsb, alsb, (long)TOK * FFNH, slots + 4);
    quantize_w<<<gs((long)HID * FFNH), 256, 0, stream>>>(w3_w[i], wq, (long)HID * FFNH, slots + 5);
    {
      dim3 g(HID / QBN, TOK / QBM, 1);
      gemm_iu8_dual<<<g, 256, 0, stream>>>(amsb, alsb, wq, cm2, cl2,
                                           TOK, HID, FFNH,
                                           (int*)(slots + 6), (int*)(slots + 7));
    }
    bitserial_finalize<<<gs((long)TOK * HID), 256, 0, stream>>>(
        cm2, cl2, w3_b[i], nullptr, hbuf, mb, 5 * HID,
        TOK, HID, slots + 4, slots + 5, (const int*)(slots + 6), (const int*)(slots + 7));
  }

  gemm(csilu, fada_w, fada_b, mb, Bx, 2 * HID, HID, 1, 0, 0, 0, 1.f, 0);
  rms_modulate<<<TOK, 256, 0, stream>>>(hbuf, fn_w, mb, 2 * HID, 0, HID, xm);
  gemm(xm, flin_w, flin_b, tmp, TOK, PATCH * PATCH * CIN, HID, 1, 0, 0, 0, 1.f, 0);
  unpatchify<<<(int)(((long)Bx * CIN * IMGSZ * IMGSZ + 255) / 256), 256, 0, stream>>>(
      tmp, (float*)d_out);
}
